// GraphSage_79216376807521
// MI455X (gfx1250) — matrix operand order repaired and checked by reading, not compiled, not run on hardware
//
#include <hip/hip_runtime.h>

// ---------------------------------------------------------------------------
// GraphSAGE 2-hop forward for MI455X (gfx1250, wave32, WMMA).
// All four matmuls use v_wmma_f32_16x16x32_f16 with fp32 accumulation.
// fp32 diffusion matrices are converted to f16 inline (single HBM pass,
// 256 MB stream = the roofline floor). B operands are gathered/transposed to
// f16 [N][K] so WMMA B-fragments are contiguous b128 loads (L2-resident).
// Main K-loop is software-pipelined: next-step A/B fragments are fetched
// before the current WMMAs issue; K-tail is peeled out of the hot loop.
// ---------------------------------------------------------------------------

typedef __attribute__((ext_vector_type(16))) _Float16 v16h;
typedef __attribute__((ext_vector_type(8)))  _Float16 v8h;
typedef __attribute__((ext_vector_type(8)))  float    v8f;
typedef __attribute__((ext_vector_type(4)))  float    f32x4;

// BT[d][i] = f16(feats[idx[i]][d])   (idx==nullptr -> identity, used for W^T)
// Pads i in [S, Spad) with zeros so the GEMM K-loop can run over Kb.
template <typename TIN>
__global__ __launch_bounds__(256) void gather_transpose_f16(
    const TIN* __restrict__ feats, const int* __restrict__ idx,
    _Float16* __restrict__ BT, int S, int Spad)
{
  const int i = blockIdx.x * blockDim.x + threadIdx.x;
  const int d = blockIdx.y;  // 0..127
  if (i >= Spad) return;
  _Float16 v = (_Float16)0.0f;
  if (i < S) {
    const int r = idx ? idx[i] : i;
    v = (_Float16)feats[(size_t)r * 128 + d];
  }
  BT[(size_t)d * Spad + i] = v;
}

// out[i][coloff + d] = f16(feats[idx[i]][d])  (row-major gather, D = 128)
template <typename TIN>
__global__ __launch_bounds__(256) void gather_rows_f16(
    const TIN* __restrict__ feats, const int* __restrict__ idx,
    _Float16* __restrict__ out, int ldo, int coloff, int M)
{
  const int t = blockIdx.x * blockDim.x + threadIdx.x;
  const int i = t >> 7;
  const int d = t & 127;
  if (i >= M) return;
  const int r = idx[i];
  out[(size_t)i * ldo + coloff + d] = (_Float16)feats[(size_t)r * 128 + d];
}

// C[M x 128] = relu?(A[M x Ktrue] @ B)   with B given as BT[128][Kb] (f16).
// A is fp32 (converted inline to f16) or f16. Kb is a multiple of 32; BT rows
// are zero-padded for k in [Ktrue, Kb).
//
// Block: 256 threads = 8 waves, tile M=32 x N=128 (waves: 2 in M x 4 in N,
// each wave owns a 16x32 slab = two 16x16 f32 accumulator tiles).
template <bool A_IS_F32, bool RELU, bool OUT_F16>
__global__ __launch_bounds__(256) void wmma_gemm_n128(
    const void* __restrict__ Araw, int lda, int Ktrue,
    const _Float16* __restrict__ BT, int Kb,
    void* __restrict__ Craw, int ldc, int M)
{
  const int lane = threadIdx.x & 31;
  const int wave = threadIdx.x >> 5;
  const int hsel = lane >> 4;   // 0: lanes 0-15, 1: lanes 16-31
  const int l16  = lane & 15;

  const int m_base = blockIdx.x * 32 + (wave >> 2) * 16;
  const int n_base = (wave & 3) * 32;

  // A-row this lane feeds (ISA 7.12.2: 16-bit A, lane%16 = M row). Clamp for
  // the M tail; stores are masked below (WMMA needs EXEC all-ones, so clamp
  // addresses instead of predicating loads).
  int arow = m_base + l16;
  if (arow > M - 1) arow = M - 1;

  const float*    Af = (const float*)Araw    + (size_t)arow * lda;
  const _Float16* Ah = (const _Float16*)Araw + (size_t)arow * lda;
  const _Float16* B0 = BT + (size_t)(n_base + l16) * Kb;       // tile t=0 column
  const _Float16* B1 = BT + (size_t)(n_base + 16 + l16) * Kb;  // tile t=1 column

  // ---- fragment loaders -------------------------------------------------
  // A layout: lane<16 holds K = kk..kk+7 and kk+16..kk+23;
  //           lane>=16 holds K = kk+8..kk+15 and kk+24..kk+31.
  auto loadA_fast = [&](int kk) -> v16h {
    v16h a;
    const int k0 = kk + hsel * 8;
    if (A_IS_F32) {
      f32x4 a0 = *(const f32x4*)(Af + k0);
      f32x4 a1 = *(const f32x4*)(Af + k0 + 4);
      f32x4 a2 = *(const f32x4*)(Af + k0 + 16);
      f32x4 a3 = *(const f32x4*)(Af + k0 + 20);
#pragma unroll
      for (int e = 0; e < 4; ++e) {
        a[e]      = (_Float16)a0[e];
        a[e + 4]  = (_Float16)a1[e];
        a[e + 8]  = (_Float16)a2[e];
        a[e + 12] = (_Float16)a3[e];
      }
    } else {
      v8h lo = *(const v8h*)(Ah + k0);
      v8h hi = *(const v8h*)(Ah + k0 + 16);
#pragma unroll
      for (int e = 0; e < 8; ++e) { a[e] = lo[e]; a[e + 8] = hi[e]; }
    }
    return a;
  };
  // B layout: same K order as A (lane<16: K=kk..kk+7, kk+16..kk+23;
  // lane>=16: K=kk+8..kk+15, kk+24..kk+31) -> two 16B loads.
  auto loadB = [&](const _Float16* Bcol, int kk) -> v16h {
    v8h lo = *(const v8h*)(Bcol + kk + hsel * 8);
    v8h hi = *(const v8h*)(Bcol + kk + hsel * 8 + 16);
    v16h b;
#pragma unroll
    for (int e = 0; e < 8; ++e) { b[e] = lo[e]; b[e + 8] = hi[e]; }
    return b;
  };

  v8f acc0, acc1;
#pragma unroll
  for (int r = 0; r < 8; ++r) { acc0[r] = 0.0f; acc1[r] = 0.0f; }

  const int Kmain = Ktrue & ~31;  // fast-path K extent (multiple of 32)

  // ---- main loop: software-pipelined, branch-free body ------------------
  if (Kmain > 0) {
    v16h a  = loadA_fast(0);
    v16h b0 = loadB(B0, 0);
    v16h b1 = loadB(B1, 0);
    for (int kk = 32; kk < Kmain; kk += 32) {
      if (A_IS_F32)  // stream-ahead prefetch on the HBM-resident operand
        __builtin_prefetch(Af + kk + 256, 0, 3);
      v16h a_n  = loadA_fast(kk);
      v16h b0_n = loadB(B0, kk);
      v16h b1_n = loadB(B1, kk);
      acc0 = __builtin_amdgcn_wmma_f32_16x16x32_f16(
          false, a, false, b0, (short)0, acc0, false, false);
      acc1 = __builtin_amdgcn_wmma_f32_16x16x32_f16(
          false, a, false, b1, (short)0, acc1, false, false);
      a = a_n; b0 = b0_n; b1 = b1_n;
    }
    acc0 = __builtin_amdgcn_wmma_f32_16x16x32_f16(
        false, a, false, b0, (short)0, acc0, false, false);
    acc1 = __builtin_amdgcn_wmma_f32_16x16x32_f16(
        false, a, false, b1, (short)0, acc1, false, false);
  }

  // ---- peeled K tail (only dif_mat_1, K=3500): masked A, zero-padded B --
  if (Kmain < Kb) {
    const int kk = Kmain;
    v16h a;
    const int k0 = kk + hsel * 8;
#pragma unroll
    for (int e = 0; e < 8; ++e) {
      const int ka = k0 + e;
      const int kb = k0 + 16 + e;
      if (A_IS_F32) {
        a[e]     = (ka < Ktrue) ? (_Float16)Af[ka] : (_Float16)0.0f;
        a[e + 8] = (kb < Ktrue) ? (_Float16)Af[kb] : (_Float16)0.0f;
      } else {
        a[e]     = (ka < Ktrue) ? Ah[ka] : (_Float16)0.0f;
        a[e + 8] = (kb < Ktrue) ? Ah[kb] : (_Float16)0.0f;
      }
    }
    v16h b0 = loadB(B0, kk);
    v16h b1 = loadB(B1, kk);
    acc0 = __builtin_amdgcn_wmma_f32_16x16x32_f16(
        false, a, false, b0, (short)0, acc0, false, false);
    acc1 = __builtin_amdgcn_wmma_f32_16x16x32_f16(
        false, a, false, b1, (short)0, acc1, false, false);
  }

  // ---- store (C layout: VGPR r, lanes<16 -> M=r, lanes>=16 -> M=8+r) ----
  const int row0 = m_base + hsel * 8;
#pragma unroll
  for (int t = 0; t < 2; ++t) {
    const v8f& acc = t ? acc1 : acc0;
    const int n = n_base + t * 16 + l16;
#pragma unroll
    for (int r = 0; r < 8; ++r) {
      const int row = row0 + r;
      if (row < M) {
        float v = acc[r];
        if (RELU) v = v > 0.0f ? v : 0.0f;
        if (OUT_F16) ((_Float16*)Craw)[(size_t)row * ldc + n] = (_Float16)v;
        else         ((float*)Craw)[(size_t)row * ldc + n]    = v;
      }
    }
  }
}

extern "C" void kernel_launch(void* const* d_in, const int* in_sizes, int n_in,
                              void* d_out, int out_size, void* d_ws, size_t ws_size,
                              hipStream_t stream)
{
  const float* src_nodes = (const float*)d_in[0];
  const int*   s2s1      = (const int*)d_in[1];   // [S1] hop-1 src indices
  const int*   s2s2      = (const int*)d_in[2];   // [S2] hop-2 src indices
  const int*   s2d1      = (const int*)d_in[3];   // [M1] hop-1 dst indices
  const int*   s2d2      = (const int*)d_in[4];   // [M2] hop-2 dst indices
  const float* dif1      = (const float*)d_in[5]; // [M1 x S1]
  const float* dif2      = (const float*)d_in[6]; // [M2 x S2]
  const float* w1        = (const float*)d_in[7]; // [256 x 128]
  const float* w2        = (const float*)d_in[8]; // [256 x 128]

  const int D  = 128;
  const int S1 = in_sizes[1], S2 = in_sizes[2];
  const int M1 = in_sizes[3], M2 = in_sizes[4];
  const int Kc = 2 * D;                        // 256 (concat K)
  const int S1p = (S1 + 31) & ~31;             // 3520
  const int S2p = (S2 + 31) & ~31;             // 16000

  // Workspace carve-out (256B aligned), ~8.7 MB total (all L2-resident).
  char* ws = (char*)d_ws;
  auto carve = [&](size_t bytes) -> char* {
    char* p = ws; ws += (bytes + 255) & ~(size_t)255; return p;
  };
  _Float16* F2T   = (_Float16*)carve((size_t)D  * S2p * 2); // B of GEMM1: [128][S2p]
  _Float16* A1cat = (_Float16*)carve((size_t)M2 * Kc  * 2); // [agg2 | dst2] f16
  _Float16* X     = (_Float16*)carve((size_t)M2 * D   * 2); // hop-2 output (relu'd)
  _Float16* F1T   = (_Float16*)carve((size_t)D  * S1p * 2); // B of GEMM3: [128][S1p]
  _Float16* A2cat = (_Float16*)carve((size_t)M1 * Kc  * 2); // [agg1 | dst1] f16
  _Float16* W1T   = (_Float16*)carve((size_t)D  * Kc  * 2); // w1^T f16 [128][256]
  _Float16* W2T   = (_Float16*)carve((size_t)D  * Kc  * 2); // w2^T f16 [128][256]
  (void)ws_size; (void)n_in; (void)out_size;

  const dim3 B256(256);

  // Weight transposes (fp32 -> f16, [K][128] -> [128][K])
  gather_transpose_f16<float><<<dim3(1, D), B256, 0, stream>>>(w1, nullptr, W1T, Kc, Kc);
  gather_transpose_f16<float><<<dim3(1, D), B256, 0, stream>>>(w2, nullptr, W2T, Kc, Kc);

  // Hop-2: gather src features (transposed) and dst features
  gather_transpose_f16<float><<<dim3((S2p + 255) / 256, D), B256, 0, stream>>>(
      src_nodes, s2s2, F2T, S2, S2p);
  gather_rows_f16<float><<<dim3((M2 * D + 255) / 256), B256, 0, stream>>>(
      src_nodes, s2d2, A1cat, Kc, D, M2);

  // GEMM1: agg2[M2 x 128] = dif2(fp32->f16 inline) @ F2  -> A1cat[:, 0:128]
  wmma_gemm_n128<true, false, true><<<dim3((M2 + 31) / 32), B256, 0, stream>>>(
      dif2, S2, S2, F2T, S2p, A1cat, Kc, M2);

  // GEMM2: X = relu([agg2|dst2] @ w1)   (f16 A, K=256)
  wmma_gemm_n128<false, true, true><<<dim3((M2 + 31) / 32), B256, 0, stream>>>(
      A1cat, Kc, Kc, W1T, Kc, X, D, M2);

  // Hop-1: gathers from X
  gather_transpose_f16<_Float16><<<dim3((S1p + 255) / 256, D), B256, 0, stream>>>(
      X, s2s1, F1T, S1, S1p);
  gather_rows_f16<_Float16><<<dim3((M1 * D + 255) / 256), B256, 0, stream>>>(
      X, s2d1, A2cat, Kc, D, M1);

  // GEMM3: agg1[M1 x 128] = dif1(fp32->f16 inline, K tail peeled) @ F1
  wmma_gemm_n128<true, false, true><<<dim3((M1 + 31) / 32), B256, 0, stream>>>(
      dif1, S1, S1, F1T, S1p, A2cat, Kc, M1);

  // GEMM4: out(fp32) = relu([agg1|dst1] @ w2)
  wmma_gemm_n128<false, true, false><<<dim3((M1 + 31) / 32), B256, 0, stream>>>(
      A2cat, Kc, Kc, W2T, Kc, d_out, D, M1);
}
